// GaussianRasterizer_17420387352700
// MI455X (gfx1250) — matrix operand order repaired and checked by reading, not compiled, not run on hardware
//
#include <hip/hip_runtime.h>

typedef __attribute__((ext_vector_type(16))) _Float16 v16h;
typedef __attribute__((ext_vector_type(8)))  float    v8f;

#define IMG_H 256
#define IMG_W 256
#define NPIX  (IMG_H * IMG_W)
#define MAXG  1024

union H2 { unsigned int u; _Float16 h[2]; };

// One wave rasterizes a 16-pixel tile against all NG Gaussians.
// Color accumulation (einsum 'np,nc->cp') runs on v_wmma_f32_16x16x32_f16:
//   A = w (16 pixels x 32 gaussians, f16), B = colors (32 gaussians x 16, cols 0..2 live),
//   C/D = f32 accumulator carried across NG/32 chunks.
__global__ __launch_bounds__(256) void gauss_raster_wmma(
    const float* __restrict__ opacity,
    const float* __restrict__ means,
    const float* __restrict__ stds,
    const float* __restrict__ rhos,
    const float* __restrict__ colors,
    const int*   __restrict__ scale_p,
    const int*   __restrict__ ratio_p,
    float* __restrict__ out,
    int NG)
{
    // Preprocessed Gaussian parameters, shared by all 8 waves of the block.
    __shared__ float4 sg0[MAXG];               // mx, my, 1/sx, 1/sy   (16 KB)
    __shared__ float4 sg1[MAXG];               // inv, 2*r*inv, opacity, 0 (16 KB)
    __shared__ unsigned int sColU[4][MAXG/2];  // packed half2 colors per channel; ch3 = zeros (8 KB)

    const int tid = threadIdx.x;
    const float sf     = (float)(*scale_p);
    const float rr     = (float)(*ratio_p);
    const float cutoff = rr * rr;

    // ---- per-block prep: hoist all per-Gaussian algebra out of the 67M-eval loop ----
    for (int n = tid; n < NG; n += 256) {
        float mx = means[2*n]   * sf;
        float my = means[2*n+1] * sf;
        float sx = stds[2*n]    * sf;
        float sy = stds[2*n+1]  * sf;
        float r  = fminf(0.999f, fmaxf(-0.999f, rhos[n]));
        float inv = 1.0f / (1.0f - r * r);
        sg0[n] = make_float4(mx, my, 1.0f / sx, 1.0f / sy);
        sg1[n] = make_float4(inv, 2.0f * r * inv, opacity[n], 0.0f);
    }
    for (int i = tid; i < NG / 2; i += 256) {
        #pragma unroll
        for (int c = 0; c < 3; ++c) {
            H2 p;
            p.h[0] = (_Float16)colors[(2*i)     * 3 + c];
            p.h[1] = (_Float16)colors[(2*i + 1) * 3 + c];
            sColU[c][i] = p.u;
        }
        sColU[3][i] = 0u;   // padded channel read by lanes N>=3 (keeps EXEC uniform)
    }
    __syncthreads();

    const int wave = tid >> 5;
    const int lane = tid & 31;
    const int m    = lane & 15;   // A row (pixel) / B & D column (channel)
    const int hi   = lane >> 4;
    const int pixBase = (blockIdx.x * 8 + wave) * 16;   // 16 consecutive pixels = one image row segment
    const float py = (float)(pixBase >> 8) + 0.5f;                 // IMG_W == 256
    const float px = (float)(pixBase & (IMG_W - 1)) + (float)m + 0.5f;
    const int nChC = (m < 3) ? m : 3;                               // clamp to zero channel

    v8f acc = {};
    const int nChunks = NG >> 5;
    for (int ch = 0; ch < nChunks; ++ch) {
        const int kb = ch << 5;
        v16h A, B;
        #pragma unroll
        for (int v = 0; v < 8; ++v) {
            // A-matrix 16x32 f16 layout: lane m, K pair:
            const int k0 = kb + ((v >> 2) << 4) + (hi << 3) + ((v & 3) << 1);
            float w0, w1;
            {
                float4 ga = sg0[k0];
                float4 gb = sg1[k0];
                float u  = (px - ga.x) * ga.z;
                float vv = (py - ga.y) * ga.w;
                float Q  = gb.x * (u * u + vv * vv) - gb.y * (u * vv);
                // Branchless cutoff: select the multiplier, keep exp unconditional.
                float gate = (Q <= cutoff) ? gb.z : 0.0f;
                w0 = gate * __expf(-0.5f * Q);
            }
            {
                float4 ga = sg0[k0 + 1];
                float4 gb = sg1[k0 + 1];
                float u  = (px - ga.x) * ga.z;
                float vv = (py - ga.y) * ga.w;
                float Q  = gb.x * (u * u + vv * vv) - gb.y * (u * vv);
                float gate = (Q <= cutoff) ? gb.z : 0.0f;
                w1 = gate * __expf(-0.5f * Q);
            }
            A[2*v]     = (_Float16)w0;
            A[2*v + 1] = (_Float16)w1;

            // B-matrix 32x16 f16 layout: lane column N=m, K pair k = kb + 16*(v>>2) + 8*hi + 2*(v&3)
            // -> packed pair index kb/2 + 8*(v>>2) + 4*hi + (v&3) (one dword per register).
            H2 cb;
            cb.u = sColU[nChC][(kb >> 1) + ((v >> 2) << 3) + (hi << 2) + (v & 3)];
            B[2*v]     = cb.h[0];
            B[2*v + 1] = cb.h[1];
        }
        // D = A x B + C, f32 accumulate; EXEC is all-ones here (uniform control flow).
        acc = __builtin_amdgcn_wmma_f32_16x16x32_f16(
                  /*neg_a=*/false, A, /*neg_b=*/false, B,
                  /*c_mod=*/(short)0, acc, /*reuse_a=*/false, /*reuse_b=*/false);
    }

    // D layout: lane l, reg v -> pixel = pixBase + 8*(l>>4) + v, channel = l&15.
    if (m < 3) {
        #pragma unroll
        for (int v = 0; v < 8; ++v) {
            out[m * NPIX + pixBase + (hi << 3) + v] = acc[v];
        }
    }
}

extern "C" void kernel_launch(void* const* d_in, const int* in_sizes, int n_in,
                              void* d_out, int out_size, void* d_ws, size_t ws_size,
                              hipStream_t stream) {
    const float* opacity = (const float*)d_in[0];
    const float* means   = (const float*)d_in[1];
    const float* stds    = (const float*)d_in[2];
    const float* rhos    = (const float*)d_in[3];
    const float* colors  = (const float*)d_in[4];
    const int*   scale_p = (const int*)d_in[7];   // scale_factor
    const int*   ratio_p = (const int*)d_in[8];   // raster_ratio
    float* out = (float*)d_out;

    const int NG = in_sizes[0];                   // 1024 Gaussians (multiple of 32)
    const int tilesPerBlock = 8;                  // 8 waves x 16 pixels
    dim3 grid(NPIX / (16 * tilesPerBlock));       // 512 blocks
    gauss_raster_wmma<<<grid, 256, 0, stream>>>(opacity, means, stds, rhos, colors,
                                                scale_p, ratio_p, out, NG);
}
